// EvoVisionTransformer_36352603193527
// MI455X (gfx1250) — compile-verified, matrix operand order checked by reading, not run on hardware
//
#include <hip/hip_runtime.h>
#include <math.h>

typedef __bf16 bf16;
typedef __attribute__((ext_vector_type(16))) __bf16 v16bf;
typedef __attribute__((ext_vector_type(8)))  float  v8f;

#define C_DIM 768
#define NHEAD 12
#define HDIM  64
#define NTOK  197
#define A0STRIDE 224

__device__ __forceinline__ int wkmap(int e, int lane) {
  // 16-bit A/B fragment K mapping per CDNA5 ISA 7.12.2 (16x32 bf16):
  // element e (0..15): K = (e>=8?16:0) + 8*(lane>=16) + (e&7)
  return ((e & 8) << 1) + ((lane >> 4) << 3) + (e & 7);
}
__device__ __forceinline__ bf16 tobf(float f) { return (bf16)f; }

__device__ __forceinline__ float wred_sum(float v) {
  for (int m = 16; m > 0; m >>= 1) v += __shfl_xor(v, m, 32);
  return v;
}
__device__ __forceinline__ float wred_max(float v) {
  for (int m = 16; m > 0; m >>= 1) v = fmaxf(v, __shfl_xor(v, m, 32));
  return v;
}

// ---------------------------------------------------------------------------
// Generic bf16-WMMA GEMM: out(MxN) = act(A(MxK) @ W(KxN) + bias) [+ res]
// Compile-time N/K -> strided weight loads become base+imm-offset vmem ops,
// static K-loop trip count, no runtime epilogue branches.
// LDS: As row-major (m x k), Bs TRANSPOSED (n x k) so both fragment types
// are two contiguous 8x-bf16 runs per lane -> ds_load_b128.
// ---------------------------------------------------------------------------
template <int N, int K, int ACT, bool RES>
__global__ __launch_bounds__(256) void gemm_bf16_wmma(
    const float* __restrict__ A, const float* __restrict__ W,
    const float* __restrict__ bias, const float* __restrict__ res,
    float* __restrict__ out, int M)
{
  __shared__ bf16 As[64][72];   // [m][k]  pitch 144B -> conflict-free frags
  __shared__ bf16 Bs[64][72];   // [n][k]  (transposed)
  const int tid  = threadIdx.x;
  const int lane = tid & 31;
  const int wave = tid >> 5;
  const int wm = wave >> 2;          // 0..1 : 32-row strip
  const int wn = wave & 3;           // 0..3 : 16-col strip
  const int bm = blockIdx.x * 64;
  const int bn = blockIdx.y * 64;
  v8f acc0 = {}; v8f acc1 = {};

  const int ar = tid >> 2;           // A stage: row 0..63
  const int ac = (tid & 3) * 16;     // A stage: col group
  const int nn = tid & 63;           // B stage: n within tile
  const int kg = (tid >> 6) * 16;    // B stage: k group 0,16,32,48
  const bool inN = (N % 64 == 0) || ((bn + nn) < N);
  const bool inM = (bm + ar) < M;
  const float* ap0 = A + (size_t)(bm + ar) * K + ac;
  const float* wp0 = W + (size_t)kg * N + bn + nn;

  for (int kk = 0; kk < K; kk += 64) {
    { // ---- stage A tile: coalesced float4 loads, packed bf16 stores ----
      if (inM) {
        const float4* ap = reinterpret_cast<const float4*>(ap0 + kk);
        #pragma unroll
        for (int i = 0; i < 4; i++) {
          float4 f = ap[i];
          As[ar][ac + 4*i + 0] = tobf(f.x);
          As[ar][ac + 4*i + 1] = tobf(f.y);
          As[ar][ac + 4*i + 2] = tobf(f.z);
          As[ar][ac + 4*i + 3] = tobf(f.w);
        }
      } else {
        #pragma unroll
        for (int i = 0; i < 16; i++) As[ar][ac + i] = tobf(0.f);
      }
    }
    { // ---- stage B tile transposed: constant imm offsets i*N ----
      const float* wp = wp0 + (size_t)kk * N;
      if (inN) {
        #pragma unroll
        for (int i = 0; i < 16; i++)
          Bs[nn][kg + i] = tobf(wp[(size_t)i * N]);
      } else {
        #pragma unroll
        for (int i = 0; i < 16; i++) Bs[nn][kg + i] = tobf(0.f);
      }
    }
    if (kk + 64 < K) { // speculative prefetch of next K tile
      __builtin_prefetch(ap0 + kk + 64, 0, 1);
      __builtin_prefetch(wp0 + (size_t)(kk + 64) * N, 0, 1);
    }
    __syncthreads();
    #pragma unroll
    for (int ks = 0; ks < 64; ks += 32) {
      v16bf a0, a1, bb;
      #pragma unroll
      for (int e = 0; e < 16; e++) {
        int kq = ks + wkmap(e, lane);
        a0[e] = As[wm * 32 +      (lane & 15)][kq];
        a1[e] = As[wm * 32 + 16 + (lane & 15)][kq];
        bb[e] = Bs[wn * 16 +      (lane & 15)][kq];
      }
      acc0 = __builtin_amdgcn_wmma_f32_16x16x32_bf16(false, a0, false, bb,
                                                     (short)0, acc0, false, false);
      acc1 = __builtin_amdgcn_wmma_f32_16x16x32_bf16(false, a1, false, bb,
                                                     (short)0, acc1, false, false);
    }
    __syncthreads();
  }

  const int col = bn + wn * 16 + (lane & 15);
  if ((N % 64 == 0) || col < N) {
    float bv = bias[col];
    #pragma unroll
    for (int v = 0; v < 8; v++) {
      int row0 = bm + wm * 32 + v + ((lane >> 4) << 3);
      int row1 = row0 + 16;
      float x0 = acc0[v] + bv;
      float x1 = acc1[v] + bv;
      if (ACT == 1) {
        x0 = 0.5f * x0 * (1.f + erff(x0 * 0.70710678f));
        x1 = 0.5f * x1 * (1.f + erff(x1 * 0.70710678f));
      }
      if (RES) {
        if (row0 < M) out[(size_t)row0 * N + col] = x0 + res[(size_t)row0 * N + col];
        if (row1 < M) out[(size_t)row1 * N + col] = x1 + res[(size_t)row1 * N + col];
      } else {
        if (row0 < M) out[(size_t)row0 * N + col] = x0;
        if (row1 < M) out[(size_t)row1 * N + col] = x1;
      }
    }
  }
}

// ---------------------------------------------------------------------------
// Fused attention: per (16-query tile, head, batch), NT compile-time.
// Phase 1: scores = Q@K^T (K staged token-major in KVs).
// Phase 2: softmax in fp32 LDS; V re-staged TRANSPOSED into the same KVs
//          region (K is dead), so P@V B-fragments are contiguous too.
// ---------------------------------------------------------------------------
#define KS(r, c) KVs[(r) * 72 + (c)]     // K: [token][hd]   224 x 72
#define VS(n, t) KVs[(n) * 232 + (t)]    // V: [hd][token]   64 x 232
template <int NT>
__global__ __launch_bounds__(256) void attn_kernel(
    const float* __restrict__ qk, const float* __restrict__ vbuf,
    float* __restrict__ obuf, float* __restrict__ attn0)
{
  constexpr int NtPad = (NT + 31) & ~31;
  __shared__ bf16  Qs[16][72];
  __shared__ bf16  KVs[224 * 72];
  __shared__ float Ss[16][236];
  const int tid = threadIdx.x, lane = tid & 31, wave = tid >> 5;
  const int mt = blockIdx.x, h = blockIdx.y, b = blockIdx.z;
  const int qrow0 = mt * 16;

  { // ---- stage Q (16 x 64): float4 loads, guard hoisted per row ----
    int rr = tid >> 4, c4 = (tid & 15) * 4;
    int gq = qrow0 + rr;
    if (gq < NT) {
      float4 f = *reinterpret_cast<const float4*>(
          qk + (size_t)(b * NT + gq) * (2 * C_DIM) + h * HDIM + c4);
      Qs[rr][c4 + 0] = tobf(f.x); Qs[rr][c4 + 1] = tobf(f.y);
      Qs[rr][c4 + 2] = tobf(f.z); Qs[rr][c4 + 3] = tobf(f.w);
    } else {
      Qs[rr][c4 + 0] = tobf(0.f); Qs[rr][c4 + 1] = tobf(0.f);
      Qs[rr][c4 + 2] = tobf(0.f); Qs[rr][c4 + 3] = tobf(0.f);
    }
  }
  { // ---- stage K token-major (NtPad x 64) ----
    int c4 = (tid & 15) * 4;
    #pragma unroll
    for (int rr0 = 0; rr0 < NtPad; rr0 += 16) {
      int rr = rr0 + (tid >> 4);
      if (rr < NT) {
        float4 f = *reinterpret_cast<const float4*>(
            qk + (size_t)(b * NT + rr) * (2 * C_DIM) + C_DIM + h * HDIM + c4);
        KS(rr, c4 + 0) = tobf(f.x); KS(rr, c4 + 1) = tobf(f.y);
        KS(rr, c4 + 2) = tobf(f.z); KS(rr, c4 + 3) = tobf(f.w);
      } else {
        KS(rr, c4 + 0) = tobf(0.f); KS(rr, c4 + 1) = tobf(0.f);
        KS(rr, c4 + 2) = tobf(0.f); KS(rr, c4 + 3) = tobf(0.f);
      }
    }
  }
  __syncthreads();

  // ---- scores = q @ k^T * scale ----
  constexpr int nTiles = NtPad >> 4;
  for (int nt = wave; nt < nTiles; nt += 8) {
    v8f acc = {};
    #pragma unroll
    for (int ks = 0; ks < 64; ks += 32) {
      v16bf a, bb;
      #pragma unroll
      for (int e = 0; e < 16; e++) {
        int kq = ks + wkmap(e, lane);
        a[e]  = Qs[lane & 15][kq];
        bb[e] = KS(nt * 16 + (lane & 15), kq);
      }
      acc = __builtin_amdgcn_wmma_f32_16x16x32_bf16(false, a, false, bb,
                                                    (short)0, acc, false, false);
    }
    #pragma unroll
    for (int v = 0; v < 8; v++)
      Ss[v + ((lane >> 4) << 3)][nt * 16 + (lane & 15)] = acc[v] * 0.125f; // HD^-0.5
  }
  __syncthreads();   // all K reads done; KVs may be re-used for V

  { // ---- stage V transposed [hd][token] into KVs (K region dead) ----
    constexpr int G = NtPad >> 2;        // tokens per thread
    int cc = tid & 63;
    int t0 = (tid >> 6) * G;
    int tl = t0 + G;
    int tReal = (tl < NT) ? tl : NT;
    for (int t = t0; t < tReal; t++)
      VS(cc, t) = tobf(vbuf[(size_t)(b * NT + t) * C_DIM + h * HDIM + cc]);
    for (int t = (t0 > NT ? t0 : NT); t < tl; t++)
      VS(cc, t) = tobf(0.f);
  }

  // ---- softmax per row (2 rows per wave) ----
  for (int rr = wave * 2; rr < wave * 2 + 2; rr++) {
    int gq = qrow0 + rr;
    if (gq < NT) {
      float m = -3.4e38f;
      for (int j = lane; j < NT; j += 32) m = fmaxf(m, Ss[rr][j]);
      m = wred_max(m);
      float s = 0.f;
      for (int j = lane; j < NT; j += 32) { float e = expf(Ss[rr][j] - m); Ss[rr][j] = e; s += e; }
      s = wred_sum(s);
      float inv = 1.f / s;
      for (int j = lane; j < NtPad; j += 32) Ss[rr][j] = (j < NT) ? Ss[rr][j] * inv : 0.f;
      if (gq == 0)
        for (int j = lane; j < NT; j += 32)
          attn0[(size_t)(b * NHEAD + h) * A0STRIDE + j] = Ss[rr][j];
    } else {
      for (int j = lane; j < NtPad; j += 32) Ss[rr][j] = 0.f;
    }
  }
  __syncthreads();

  // ---- out = attn @ V  (waves 0..3 each own 16 of the 64 head-dim cols) ----
  if (wave < 4) {
    v8f acc = {};
    #pragma unroll
    for (int kt = 0; kt < NtPad; kt += 32) {
      v16bf a, bb;
      #pragma unroll
      for (int e = 0; e < 16; e++) {
        int kq = kt + wkmap(e, lane);
        a[e]  = tobf(Ss[lane & 15][kq]);
        bb[e] = VS(wave * 16 + (lane & 15), kq);
      }
      acc = __builtin_amdgcn_wmma_f32_16x16x32_bf16(false, a, false, bb,
                                                    (short)0, acc, false, false);
    }
    #pragma unroll
    for (int v = 0; v < 8; v++) {
      int gq = qrow0 + v + ((lane >> 4) << 3);
      if (gq < NT)
        obuf[(size_t)(b * NT + gq) * C_DIM + h * HDIM + wave * 16 + (lane & 15)] = acc[v];
    }
  }
}

// ---------------------------------------------------------------------------
// LayerNorm: one wave per 768-wide row.
// ---------------------------------------------------------------------------
__global__ __launch_bounds__(256) void ln_kernel(
    const float* __restrict__ in, long istr, float* __restrict__ out, long ostr,
    const float* __restrict__ g, const float* __restrict__ bparam, int M)
{
  const int wave = threadIdx.x >> 5, lane = threadIdx.x & 31;
  const int row = blockIdx.x * 8 + wave;
  if (row >= M) return;
  const float* p = in + (long)row * istr;
  float s = 0.f;
  for (int c = lane; c < C_DIM; c += 32) s += p[c];
  s = wred_sum(s);
  float mean = s * (1.f / C_DIM);
  float v = 0.f;
  for (int c = lane; c < C_DIM; c += 32) { float d = p[c] - mean; v += d * d; }
  v = wred_sum(v);
  float rstd = rsqrtf(v * (1.f / C_DIM) + 1e-6f);
  float* q = out + (long)row * ostr;
  for (int c = lane; c < C_DIM; c += 32)
    q[c] = (p[c] - mean) * rstd * g[c] + bparam[c];
}

// ------------------------- small utility kernels ---------------------------
__global__ void im2col_kernel(const float* __restrict__ img, float* __restrict__ patches) {
  int p = blockIdx.x, b = blockIdx.y;
  int gy = p / 14, gx = p % 14;
  for (int c = threadIdx.x; c < 768; c += blockDim.x) {
    int ch = c >> 8, py = (c >> 4) & 15, px = c & 15;
    patches[(size_t)(b * 196 + p) * 768 + c] =
        img[(size_t)((b * 3 + ch) * 224 + gy * 16 + py) * 224 + gx * 16 + px];
  }
}

__global__ void assemble_x(const float* __restrict__ pe, const float* __restrict__ cls_tok,
                           const float* __restrict__ pos, float* __restrict__ x) {
  int tt = blockIdx.x, b = blockIdx.y;
  for (int c = threadIdx.x; c < 768; c += blockDim.x) {
    float v = (tt == 0) ? cls_tok[c] : pe[(size_t)(b * 196 + tt - 1) * 768 + c];
    x[(size_t)(b * NTOK + tt) * 768 + c] = v + pos[(size_t)tt * 768 + c];
  }
}

__global__ void rowupdate(const float* __restrict__ attn0, float* __restrict__ cls,
                          int Nt, int mode) {
  int b = blockIdx.x;
  int L = Nt - 1;
  for (int j = threadIdx.x; j < L; j += blockDim.x) {
    float s = 0.f;
    for (int h = 0; h < NHEAD; h++) s += attn0[(size_t)(b * NHEAD + h) * A0STRIDE + 1 + j];
    if (mode == 0)      cls[b * 196 + j] = s;
    else if (mode == 1) cls[b * 196 + j] = 0.5f * cls[b * 196 + j] + 0.5f * s;
    else if (j < 98)    cls[b * 196 + j] = 0.5f * cls[b * 196 + j] + 0.5f * s;
  }
}

__global__ void rank_kernel(const float* __restrict__ cls, float* __restrict__ sorted,
                            int* __restrict__ idx) {
  int b = blockIdx.x, j = threadIdx.x;
  if (j < 196) {
    float vj = cls[b * 196 + j];
    int r = 0;
    for (int i = 0; i < 196; i++) {
      float vi = cls[b * 196 + i];
      r += (vi > vj) || (vi == vj && i < j);   // stable descending rank
    }
    sorted[b * 196 + r] = vj;
    idx[b * 196 + r] = j;
  }
}

__global__ void copyf(const float* __restrict__ s, float* __restrict__ d, int n) {
  int i = blockIdx.x * blockDim.x + threadIdx.x;
  if (i < n) d[i] = s[i];
}

__global__ void rep_kernel(const float* __restrict__ x, const float* __restrict__ clsS,
                           const int* __restrict__ idx, float* __restrict__ rep) {
  int b = blockIdx.x;
  float ssum = 0.f;
  for (int r = 98; r < 196; r++) ssum += clsS[b * 196 + r];
  float inv = 1.f / ssum;
  for (int c = threadIdx.x; c < 768; c += blockDim.x) {
    float acc = 0.f;
    for (int r = 98; r < 196; r++) {
      int tk = idx[b * 196 + r];
      acc += clsS[b * 196 + r] * x[(size_t)(b * NTOK + 1 + tk) * 768 + c];
    }
    rep[b * 768 + c] = acc * inv;
  }
}

__global__ void build_xs(const float* __restrict__ x, const int* __restrict__ idx,
                         const float* __restrict__ rep, float* __restrict__ xs) {
  int tt = blockIdx.x, b = blockIdx.y;
  const float* src;
  if (tt == 0)       src = x + (size_t)(b * NTOK) * 768;
  else if (tt <= 98) src = x + (size_t)(b * NTOK + 1 + idx[b * 196 + tt - 1]) * 768;
  else               src = rep + (size_t)b * 768;
  for (int c = threadIdx.x; c < 768; c += blockDim.x)
    xs[(size_t)(b * 100 + tt) * 768 + c] = src[c];
}

__global__ void finalize_prune(const float* __restrict__ xs, const float* __restrict__ xold,
                               const int* __restrict__ idx, const float* __restrict__ rep,
                               float* __restrict__ xnew) {
  int tt = blockIdx.x, b = blockIdx.y;
  for (int c = threadIdx.x; c < 768; c += blockDim.x) {
    float v;
    if (tt < 99) v = xs[(size_t)(b * 100 + tt) * 768 + c];
    else {
      int tk = idx[b * 196 + 98 + (tt - 99)];
      v = xold[(size_t)(b * NTOK + 1 + tk) * 768 + c]
        + 0.5f * (xs[(size_t)(b * 100 + 99) * 768 + c] - rep[(size_t)b * 768 + c]);
    }
    xnew[(size_t)(b * NTOK + tt) * 768 + c] = v;
  }
}

// ---------------------------------------------------------------------------
extern "C" void kernel_launch(void* const* d_in, const int* in_sizes, int n_in,
                              void* d_out, int out_size, void* d_ws, size_t ws_size,
                              hipStream_t stream) {
  (void)in_sizes; (void)n_in; (void)out_size; (void)ws_size;
  const float* x_img   = (const float*)d_in[0];
  const float* cls_tok = (const float*)d_in[1];
  const float* pos     = (const float*)d_in[2];
  const float* patch_w = (const float*)d_in[3];
  const float* patch_b = (const float*)d_in[4];
  const float* ln1_g   = (const float*)d_in[5];
  const float* ln1_b   = (const float*)d_in[6];
  const float* qk_w    = (const float*)d_in[7];
  const float* qk_b    = (const float*)d_in[8];
  const float* v_w     = (const float*)d_in[9];
  const float* v_b     = (const float*)d_in[10];
  const float* proj_w  = (const float*)d_in[11];
  const float* proj_b  = (const float*)d_in[12];
  const float* ln2_g   = (const float*)d_in[13];
  const float* ln2_b   = (const float*)d_in[14];
  const float* fc1_w   = (const float*)d_in[15];
  const float* fc1_b   = (const float*)d_in[16];
  const float* fc2_w   = (const float*)d_in[17];
  const float* fc2_b   = (const float*)d_in[18];
  const float* norm_g  = (const float*)d_in[19];
  const float* norm_b  = (const float*)d_in[20];
  const float* head_w  = (const float*)d_in[21];
  const float* head_b  = (const float*)d_in[22];

  float* ws = (float*)d_ws;
  size_t off = 0;
  auto alloc = [&](size_t n) { float* p = ws + off; off += n; return p; };
  const size_t SX = (size_t)32 * NTOK * 768;
  float* xb   = alloc(SX);                          // token stream (B,197,C)
  float* tb   = alloc(SX);                          // LN outputs / staging
  float* qkb  = alloc(2 * SX);                      // qk projections
  float* vb   = alloc(SX);                          // v projections
  float* ob   = alloc(SX);                          // attention output
  float* h1   = alloc((size_t)32 * NTOK * 3072);    // fc1 output / patches
  float* a0   = alloc((size_t)32 * NHEAD * A0STRIDE);
  float* cls  = alloc(32 * 196);
  float* clsS = alloc(32 * 196);
  float* rep  = alloc(32 * 768);
  float* xs   = alloc((size_t)32 * 100 * 768);
  int*   idx  = (int*)alloc(32 * 196);

  // ---- patch embed ----
  im2col_kernel<<<dim3(196, 32), 256, 0, stream>>>(x_img, h1);
  {
    int M = 32 * 196;
    gemm_bf16_wmma<768, 768, 0, false><<<dim3((M + 63) / 64, 12), 256, 0, stream>>>(
        h1, patch_w, patch_b, nullptr, ob, M);
  }
  assemble_x<<<dim3(NTOK, 32), 256, 0, stream>>>(ob, cls_tok, pos, xb);

  auto run_block = [&](float* X, int Nt, int i) {
    int M = 32 * Nt;
    int GM = (M + 63) / 64;
    ln_kernel<<<(M + 7) / 8, 256, 0, stream>>>(X, 768, tb, 768, ln1_g + i * 768, ln1_b + i * 768, M);
    gemm_bf16_wmma<1536, 768, 0, false><<<dim3(GM, 24), 256, 0, stream>>>(
        tb, qk_w + (size_t)i * 768 * 1536, qk_b + i * 1536, nullptr, qkb, M);
    gemm_bf16_wmma<768, 768, 0, false><<<dim3(GM, 12), 256, 0, stream>>>(
        tb, v_w + (size_t)i * 768 * 768, v_b + i * 768, nullptr, vb, M);
    if (Nt == NTOK)
      attn_kernel<NTOK><<<dim3((NTOK + 15) / 16, NHEAD, 32), 256, 0, stream>>>(qkb, vb, ob, a0);
    else
      attn_kernel<100><<<dim3(100 / 16 + 1, NHEAD, 32), 256, 0, stream>>>(qkb, vb, ob, a0);
    gemm_bf16_wmma<768, 768, 0, true><<<dim3(GM, 12), 256, 0, stream>>>(
        ob, proj_w + (size_t)i * 768 * 768, proj_b + i * 768, X, X, M);
    ln_kernel<<<(M + 7) / 8, 256, 0, stream>>>(X, 768, tb, 768, ln2_g + i * 768, ln2_b + i * 768, M);
    gemm_bf16_wmma<3072, 768, 1, false><<<dim3(GM, 48), 256, 0, stream>>>(
        tb, fc1_w + (size_t)i * 768 * 3072, fc1_b + i * 3072, nullptr, h1, M);
    gemm_bf16_wmma<768, 3072, 0, true><<<dim3(GM, 12), 256, 0, stream>>>(
        h1, fc2_w + (size_t)i * 3072 * 768, fc2_b + i * 768, X, X, M);
  };

  for (int i = 0; i < 12; i++) {
    if (i < 4) {                        // full layers
      run_block(xb, NTOK, i);
      rowupdate<<<32, 256, 0, stream>>>(a0, cls, NTOK, i == 0 ? 0 : 1);
    } else {                            // pruned layers
      rank_kernel<<<32, 256, 0, stream>>>(cls, clsS, idx);
      copyf<<<(32 * 196 + 255) / 256, 256, 0, stream>>>(clsS, cls, 32 * 196);
      rep_kernel<<<32, 256, 0, stream>>>(xb, cls, idx, rep);
      build_xs<<<dim3(100, 32), 256, 0, stream>>>(xb, idx, rep, xs);
      run_block(xs, 100, i);
      rowupdate<<<32, 256, 0, stream>>>(a0, cls, 100, 2);
      finalize_prune<<<dim3(NTOK, 32), 256, 0, stream>>>(xs, xb, idx, rep, tb);
      hipMemcpyAsync(xb, tb, SX * sizeof(float), hipMemcpyDeviceToDevice, stream);
    }
  }

  // ---- final norm on cls tokens + classifier head ----
  ln_kernel<<<4, 256, 0, stream>>>(xb, (long)NTOK * 768, tb, 768, norm_g, norm_b, 32);
  gemm_bf16_wmma<1000, 768, 0, false><<<dim3(1, (1000 + 63) / 64), 256, 0, stream>>>(
      tb, head_w, head_b, nullptr, (float*)d_out, 32);
}
